// NystromAttention_5119601017043
// MI455X (gfx1250) — matrix operand order repaired and checked by reading, not compiled, not run on hardware
//
#include <hip/hip_runtime.h>

#define S_LEN 4096
#define DHEAD 64
#define NLM   64
#define BH_N  64
#define SCALE 0.35355339059327379f   // 64^(-1/4)

typedef __bf16 bf16;
typedef __attribute__((ext_vector_type(16))) __bf16 v16bf;
typedef __attribute__((ext_vector_type(8)))  float  v8f;
typedef __attribute__((ext_vector_type(4)))  unsigned int u32x4;
typedef __attribute__((ext_vector_type(8)))  int i32x8;
typedef __attribute__((ext_vector_type(4)))  int i32x4;

__device__ __forceinline__ bf16 tobf(float f) { return (bf16)f; }

// ---- WMMA fragment helpers -------------------------------------------------
// A-fragment (16x32 bf16, row-major, contiguous k = 16h + s, same order as frag_b_sw):
// lane: h=lane>>4, M=lane&15; VGPR v holds K = 16h + 2v + {0,1}
__device__ __forceinline__ v16bf frag_a(const bf16* a, int lda, int row0, int k0, int lane) {
  const int h = (lane >> 4) & 1;
  const int m = lane & 15;
  v16bf r;
#pragma unroll
  for (int v = 0; v < 8; ++v) {
    const int kb = k0 + (h << 4) + (v << 1);
    const bf16* p = a + (row0 + m) * lda + kb;
    r[2 * v]     = p[0];
    r[2 * v + 1] = p[1];
  }
  return r;
}

// Swizzled B storage for a 64x64 (KxN) operand: each lane's 16 elements
// contiguous -> frag load becomes two ds_load_b128.
// idx(k,n) = (kblk*4 + nblk)*256 + (n&15)*16 + (k&15)
__device__ __forceinline__ int bsw(int k, int n) {
  return ((((k >> 4) << 2) | (n >> 4)) << 8) | ((n & 15) << 4) | (k & 15);
}
__device__ __forceinline__ v16bf frag_b_sw(const bf16* b, int k0, int col0, int lane) {
  const int h = (lane >> 4) & 1;
  const int n = lane & 15;
  const bf16* p = b + ((((((k0 >> 4) + h) << 2) | (col0 >> 4)) << 8) | (n << 4));
  v16bf r;
#pragma unroll
  for (int e = 0; e < 16; ++e) r[e] = p[e];
  return r;
}

__device__ __forceinline__ v8f wmma_bf16(v16bf a, v16bf b, v8f c) {
  return __builtin_amdgcn_wmma_f32_16x16x32_bf16(false, a, false, b, (short)0, c, false, false);
}

// ---- Tensor Data Mover: DMA a [rows x 64] f32 tile (row stride 64) to LDS --
// D# per CDNA5 ISA ch.8: group0 {count=1, lds_addr, global_addr, type=2},
// group1 {data_size=4B, tensor_dim0=64, tensor_dim1=rows, tile=64 x rows,
//         tensor_dim0_stride=64}; groups 2/3 zero (<=2D tensor).
// 6-arg builtin form (clang-23 / therock-10.0 lane).
__device__ __forceinline__ void tdm_load_tile_f32(const float* gsrc, unsigned lds_off, int rows) {
  const unsigned long long ga = (unsigned long long)(uintptr_t)gsrc;
  u32x4 g0;
  g0[0] = 1u;                                                   // count=1
  g0[1] = lds_off;                                              // lds_addr
  g0[2] = (unsigned)ga;                                         // global_addr[31:0]
  g0[3] = (unsigned)((ga >> 32) & 0x01FFFFFFull) | (2u << 30);  // addr[56:32] | type=2
  i32x8 g1;
  g1[0] = 0x00020000;                   // data_size=2 -> 4 bytes
  g1[1] = (int)(64u << 16);             // tensor_dim0 = 64 (low16 @ bits 63:48)
  g1[2] = (int)((unsigned)rows << 16);  // tensor_dim1 low16 @ bits 95:80
  g1[3] = (int)(64u << 16);             // tile_dim0 = 64 @ bits 127:112
  g1[4] = rows;                         // tile_dim1 @ bits 143:128
  g1[5] = 64;                           // tensor_dim0_stride low32
  g1[6] = 0;
  g1[7] = 0;
  i32x4 z4 = {0, 0, 0, 0};
  i32x8 z8 = {0, 0, 0, 0, 0, 0, 0, 0};
  __builtin_amdgcn_tensor_load_to_lds(g0, g1, z4, z4, z8, 0);
}
__device__ __forceinline__ unsigned lds_off_of(const void* p) {
  return (unsigned)(uintptr_t)p;   // generic LDS VA: low 32 bits = LDS offset
}

// ---------------------------------------------------------------- kernel 1
__global__ void __launch_bounds__(128) nys_pool(const float* __restrict__ Q,
                                                const float* __restrict__ K,
                                                float* __restrict__ Qlm,
                                                float* __restrict__ Klm) {
  const int m    = blockIdx.x;
  const int head = blockIdx.y;
  const int t    = threadIdx.x;
  const float* src = (t < 64) ? Q : K;
  float*       dst = (t < 64) ? Qlm : Klm;
  const int d = t & 63;
  const long base = (long)head * S_LEN * DHEAD + (long)m * 64 * DHEAD;
  float s = 0.f;
#pragma unroll 8
  for (int r = 0; r < 64; ++r) s += src[base + r * DHEAD + d];
  dst[(long)head * NLM * DHEAD + m * DHEAD + d] = s * (SCALE / 64.f);
}

// ---------------------------------------------------------------- kernel 2
__device__ void mm64_nn(float* d, const float* a, const float* b, int tid) {
  __syncthreads();
  for (int o = tid; o < 4096; o += 256) {
    const int r = o >> 6, c = o & 63;
    float s = 0.f;
#pragma unroll 8
    for (int k = 0; k < 64; ++k) s = fmaf(a[(r << 6) + k], b[(k << 6) + c], s);
    d[o] = s;
  }
  __syncthreads();
}

__global__ void __launch_bounds__(256) nys_inv(const float* __restrict__ Qlm,
                                               const float* __restrict__ Klm,
                                               float* __restrict__ INV) {
  __shared__ float Km[4096], Vm[4096], Tm[4096], Um[4096], Wm[4096];
  __shared__ float cs[64], rs[64], sc;
  const int tid  = threadIdx.x;
  const int head = blockIdx.x;
  const long off = (long)head * 4096;
  for (int o = tid; o < 4096; o += 256) { Tm[o] = Qlm[off + o]; Um[o] = Klm[off + o]; }
  __syncthreads();
  for (int o = tid; o < 4096; o += 256) {
    const int r = o >> 6, c = o & 63;
    float s = 0.f;
#pragma unroll 8
    for (int k = 0; k < 64; ++k) s = fmaf(Tm[(r << 6) + k], Um[(c << 6) + k], s);
    Km[o] = s;
  }
  __syncthreads();
  if (tid < 64) {
    float mx = -3.0e38f;
    for (int j = 0; j < 64; ++j) mx = fmaxf(mx, Km[(tid << 6) + j]);
    float sm = 0.f;
    for (int j = 0; j < 64; ++j) { float e = __expf(Km[(tid << 6) + j] - mx); Km[(tid << 6) + j] = e; sm += e; }
    const float inv = 1.f / sm;
    for (int j = 0; j < 64; ++j) Km[(tid << 6) + j] *= inv;
  }
  __syncthreads();
  if (tid < 64) {
    float c = 0.f, r = 0.f;
    for (int i = 0; i < 64; ++i) { c += Km[(i << 6) + tid]; r += Km[(tid << 6) + i]; }
    cs[tid] = c; rs[tid] = r;
  }
  __syncthreads();
  if (tid == 0) {
    float cm = 0.f, rm = 0.f;
    for (int i = 0; i < 64; ++i) { cm = fmaxf(cm, cs[i]); rm = fmaxf(rm, rs[i]); }
    sc = 1.f / (cm * rm);
  }
  __syncthreads();
  for (int o = tid; o < 4096; o += 256) { const int r = o >> 6, c = o & 63; Vm[o] = sc * Km[(c << 6) + r]; }
  for (int it = 0; it < 6; ++it) {
    mm64_nn(Tm, Km, Vm, tid);
    for (int o = tid; o < 4096; o += 256) Um[o] = (((o >> 6) == (o & 63)) ? 7.f : 0.f) - Tm[o];
    mm64_nn(Wm, Tm, Um, tid);
    for (int o = tid; o < 4096; o += 256) Um[o] = (((o >> 6) == (o & 63)) ? 15.f : 0.f) - Wm[o];
    mm64_nn(Wm, Tm, Um, tid);
    for (int o = tid; o < 4096; o += 256) Um[o] = (((o >> 6) == (o & 63)) ? 13.f : 0.f) - Wm[o];
    mm64_nn(Wm, Vm, Um, tid);
    for (int o = tid; o < 4096; o += 256) Vm[o] = 0.25f * Wm[o];
    __syncthreads();
  }
  for (int o = tid; o < 4096; o += 256) INV[off + o] = Vm[o];
}

// ---------------------------------------------------------------- kernel 3
// TDM-fed flash softmax over S for kernel_3 + (kernel_3 @ V); B2 = inv_k2 @ K3V
__global__ void __launch_bounds__(256) nys_k3v(const float* __restrict__ Qlm,
                                               const float* __restrict__ Kg,
                                               const float* __restrict__ Vg,
                                               const float* __restrict__ INV,
                                               float* __restrict__ B2) {
  __shared__ float Kst[2][4096];   // TDM staging (double buffer)
  __shared__ float Vst[2][4096];
  __shared__ float Lg[4096];
  __shared__ float Oacc[4096];
  __shared__ bf16  Qb[4096];
  __shared__ bf16  KTsw[4096];     // B operand, k=d, n=s (swizzled)
  __shared__ bf16  Vbsw[4096];     // B operand, k=s, n=d (swizzled)
  __shared__ bf16  Pb[4096];
  __shared__ float mrow[64], srow[64], crow[64];
  const int tid  = threadIdx.x;
  const int lane = tid & 31;
  const int w    = tid >> 5;
  const int head = blockIdx.x;
  const long qoff = (long)head * 4096;
  const float* Kbase = Kg + (long)head * S_LEN * DHEAD;
  const float* Vbase = Vg + (long)head * S_LEN * DHEAD;
  const int NC = S_LEN / 64;

  if (tid < 32) {   // prime chunk 0
    tdm_load_tile_f32(Kbase, lds_off_of(&Kst[0][0]), 64);
    tdm_load_tile_f32(Vbase, lds_off_of(&Vst[0][0]), 64);
  }
  for (int o = tid; o < 4096; o += 256) { Qb[o] = tobf(Qlm[qoff + o]); Oacc[o] = 0.f; }
  if (tid < 64) { mrow[tid] = -3.0e38f; srow[tid] = 0.f; }

  for (int c = 0; c < NC; ++c) {
    const int buf = c & 1;
    if (tid < 32) {
      if (c + 1 < NC) {  // issue next chunk, keep 2 in flight
        tdm_load_tile_f32(Kbase + (long)(c + 1) * 4096, lds_off_of(&Kst[buf ^ 1][0]), 64);
        tdm_load_tile_f32(Vbase + (long)(c + 1) * 4096, lds_off_of(&Vst[buf ^ 1][0]), 64);
        __builtin_amdgcn_s_wait_tensorcnt(2);
      } else {
        __builtin_amdgcn_s_wait_tensorcnt(0);
      }
    }
    __syncthreads();
    // convert + swizzle current chunk
    for (int o = tid; o < 4096; o += 256) {
      const int s = o >> 6, d = o & 63;
      KTsw[bsw(d, s)] = tobf(Kst[buf][o] * SCALE);
      Vbsw[bsw(s, d)] = tobf(Vst[buf][o]);
    }
    __syncthreads();
    // logits: Lg = Qlm (64x64) @ K_chunk^T
#pragma unroll
    for (int tt = 0; tt < 2; ++tt) {
      const int t = w + (tt << 3);
      const int ti = t >> 2, tj = t & 3;
      v8f acc = {};
      acc = wmma_bf16(frag_a(Qb, 64, ti * 16, 0,  lane), frag_b_sw(KTsw, 0,  tj * 16, lane), acc);
      acc = wmma_bf16(frag_a(Qb, 64, ti * 16, 32, lane), frag_b_sw(KTsw, 32, tj * 16, lane), acc);
      const int h = (lane >> 4) & 1, n = lane & 15;
#pragma unroll
      for (int r = 0; r < 8; ++r)
        Lg[((ti * 16 + r + (h << 3)) << 6) + tj * 16 + n] = acc[r];
    }
    __syncthreads();
    // online softmax update: 4 threads per row
    {
      const int m = tid >> 2, q = tid & 3;
      const float* Lr = &Lg[(m << 6) + (q << 4)];
      float pm = -3.0e38f;
#pragma unroll
      for (int j = 0; j < 16; ++j) pm = fmaxf(pm, Lr[j]);
      pm = fmaxf(pm, __shfl_xor(pm, 1, 4));
      pm = fmaxf(pm, __shfl_xor(pm, 2, 4));
      const float oldm = mrow[m];
      const float newm = fmaxf(oldm, pm);
      float ps = 0.f;
      bf16* Pr = &Pb[(m << 6) + (q << 4)];
#pragma unroll
      for (int j = 0; j < 16; ++j) { const float p = __expf(Lr[j] - newm); Pr[j] = tobf(p); ps += p; }
      ps += __shfl_xor(ps, 1, 4);
      ps += __shfl_xor(ps, 2, 4);
      if (q == 0) {
        const float corr = __expf(oldm - newm);
        srow[m] = srow[m] * corr + ps;
        mrow[m] = newm;
        crow[m] = corr;
      }
    }
    __syncthreads();
    for (int o = tid; o < 4096; o += 256) Oacc[o] *= crow[o >> 6];
    __syncthreads();
    // Oacc += P @ V_chunk
#pragma unroll
    for (int tt = 0; tt < 2; ++tt) {
      const int t = w + (tt << 3);
      const int ti = t >> 2, tj = t & 3;
      v8f acc = {};
      acc = wmma_bf16(frag_a(Pb, 64, ti * 16, 0,  lane), frag_b_sw(Vbsw, 0,  tj * 16, lane), acc);
      acc = wmma_bf16(frag_a(Pb, 64, ti * 16, 32, lane), frag_b_sw(Vbsw, 32, tj * 16, lane), acc);
      const int h = (lane >> 4) & 1, n = lane & 15;
#pragma unroll
      for (int r = 0; r < 8; ++r)
        Oacc[((ti * 16 + r + (h << 3)) << 6) + tj * 16 + n] += acc[r];
    }
    __syncthreads();
  }
  // K3V (normalized) into Lg; then B2 = inv_k2 @ K3V (f32)
  for (int o = tid; o < 4096; o += 256) Lg[o] = Oacc[o] / srow[o >> 6];
  __syncthreads();
  for (int o = tid; o < 4096; o += 256) Oacc[o] = INV[qoff + o];
  __syncthreads();
  for (int o = tid; o < 4096; o += 256) {
    const int r = o >> 6, cc = o & 63;
    float s = 0.f;
#pragma unroll 8
    for (int k = 0; k < 64; ++k) s = fmaf(Oacc[(r << 6) + k], Lg[(k << 6) + cc], s);
    B2[qoff + o] = s;
  }
}

// ---------------------------------------------------------------- kernel 4
// X = softmax(Q Klm^T) @ B2 ; 128 query rows per block, Q tile via TDM
__global__ void __launch_bounds__(256) nys_out(const float* __restrict__ Qg,
                                               const float* __restrict__ Klm,
                                               const float* __restrict__ B2,
                                               float* __restrict__ Out) {
  __shared__ float Fbuf[8192];     // Q tile staging, then reused as logits
  __shared__ bf16  Qs[128 * 64];
  __shared__ bf16  KlmSW[4096];    // B operand, k=d, n=m (swizzled)
  __shared__ bf16  B2sw[4096];     // B operand, k=m, n=d (swizzled)
  __shared__ bf16  Pb[128 * 64];
  const int tid  = threadIdx.x;
  const int lane = tid & 31;
  const int w    = tid >> 5;
  const int head = blockIdx.y;
  const int r0   = blockIdx.x * 128;
  const long qbase = (long)head * S_LEN * DHEAD + (long)r0 * DHEAD;
  const long lmoff = (long)head * 4096;

  if (tid < 32) {
    tdm_load_tile_f32(Qg + qbase, lds_off_of(&Fbuf[0]), 128);
  }
  for (int o = tid; o < 4096; o += 256) {
    const int m = o >> 6, d = o & 63;
    KlmSW[bsw(d, m)] = tobf(Klm[lmoff + o]);
    B2sw[bsw(m, d)]  = tobf(B2[lmoff + o]);
  }
  if (tid < 32) __builtin_amdgcn_s_wait_tensorcnt(0);
  __syncthreads();
  for (int o = tid; o < 8192; o += 256) Qs[o] = tobf(Fbuf[o] * SCALE);
  __syncthreads();
  // logits: wave w owns rows [16w, 16w+16)
#pragma unroll
  for (int tj = 0; tj < 4; ++tj) {
    v8f acc = {};
    acc = wmma_bf16(frag_a(Qs, 64, w * 16, 0,  lane), frag_b_sw(KlmSW, 0,  tj * 16, lane), acc);
    acc = wmma_bf16(frag_a(Qs, 64, w * 16, 32, lane), frag_b_sw(KlmSW, 32, tj * 16, lane), acc);
    const int h = (lane >> 4) & 1, n = lane & 15;
#pragma unroll
    for (int r = 0; r < 8; ++r)
      Fbuf[((w * 16 + r + (h << 3)) << 6) + tj * 16 + n] = acc[r];
  }
  __syncthreads();
  // row softmax: 2 threads per row
  {
    const int m = tid >> 1, q = tid & 1;
    const float* Lr = &Fbuf[(m << 6) + (q << 5)];
    float mx = -3.0e38f;
#pragma unroll
    for (int j = 0; j < 32; ++j) mx = fmaxf(mx, Lr[j]);
    mx = fmaxf(mx, __shfl_xor(mx, 1, 2));
    float sm = 0.f;
    float e[32];
#pragma unroll
    for (int j = 0; j < 32; ++j) { e[j] = __expf(Lr[j] - mx); sm += e[j]; }
    sm += __shfl_xor(sm, 1, 2);
    const float inv = 1.f / sm;
    bf16* Pr = &Pb[(m << 6) + (q << 5)];
#pragma unroll
    for (int j = 0; j < 32; ++j) Pr[j] = tobf(e[j] * inv);
  }
  __syncthreads();
  // X = P @ B2 -> global
#pragma unroll
  for (int tj = 0; tj < 4; ++tj) {
    v8f acc = {};
    acc = wmma_bf16(frag_a(Pb, 64, w * 16, 0,  lane), frag_b_sw(B2sw, 0,  tj * 16, lane), acc);
    acc = wmma_bf16(frag_a(Pb, 64, w * 16, 32, lane), frag_b_sw(B2sw, 32, tj * 16, lane), acc);
    const int h = (lane >> 4) & 1, n = lane & 15;
#pragma unroll
    for (int r = 0; r < 8; ++r)
      Out[qbase + (long)(((w * 16 + r + (h << 3)) << 6) + tj * 16 + n)] = acc[r];
  }
}

// ---------------------------------------------------------------- launcher
extern "C" void kernel_launch(void* const* d_in, const int* in_sizes, int n_in,
                              void* d_out, int out_size, void* d_ws, size_t ws_size,
                              hipStream_t stream) {
  const float* Q = (const float*)d_in[0];
  const float* K = (const float*)d_in[1];
  const float* V = (const float*)d_in[2];
  float* out = (float*)d_out;
  float* ws  = (float*)d_ws;

  float* Qlm = ws;
  float* Klm = ws + 1 * 262144;
  float* INV = ws + 2 * 262144;
  float* B2  = ws + 3 * 262144;

  nys_pool<<<dim3(NLM, BH_N), 128, 0, stream>>>(Q, K, Qlm, Klm);
  nys_inv<<<BH_N, 256, 0, stream>>>(Qlm, Klm, INV);
  nys_k3v<<<BH_N, 256, 0, stream>>>(Qlm, K, V, INV, B2);
  nys_out<<<dim3(S_LEN / 128, BH_N), 256, 0, stream>>>(Q, Klm, B2, out);
}
